// GPT_5265629904931
// MI455X (gfx1250) — matrix operand order repaired and checked by reading, not compiled, not run on hardware
//
#include <hip/hip_runtime.h>
#include <hip/hip_bf16.h>

// ---------------- model dims ----------------
#define TSEQ   1024
#define NTOK   2048      // B*T
#define DMODEL 768
#define QKV3   2304
#define NHEAD  12
#define HDIM   64
#define FFD    3072
#define NEXP   8
#define TOPK   2
#define CAPE   640
#define NVOC   50257
#define NLAYER 12

// ---------------- WMMA types ----------------
typedef __bf16 bf16_t;
typedef bf16_t bf16x8  __attribute__((ext_vector_type(8)));
typedef bf16_t bf16x16 __attribute__((ext_vector_type(16)));
typedef float  f32x8   __attribute__((ext_vector_type(8)));

#define LDT 72   // padded LDS row stride in bf16 elems (144B, 16B-aligned)

// ---------------- async global->LDS copy (CDNA5 ASYNCcnt path) ----------------
#if defined(__has_builtin)
#if __has_builtin(__builtin_amdgcn_global_load_async_to_lds_b128) && \
    __has_builtin(__builtin_amdgcn_s_wait_asynccnt)
#define USE_ASYNC_LDS 1
#endif
#endif

typedef int v4i_vec __attribute__((vector_size(4 * sizeof(int))));
typedef __attribute__((address_space(1))) v4i_vec* g_v4i_ptr;
typedef __attribute__((address_space(3))) v4i_vec* l_v4i_ptr;

// copy 16 bytes (8 bf16) from global to LDS
__device__ static inline void copy16_g2l(bf16_t* ldsdst, const bf16_t* gsrc) {
#ifdef USE_ASYNC_LDS
  __builtin_amdgcn_global_load_async_to_lds_b128(
      (g_v4i_ptr)(bf16_t*)gsrc, (l_v4i_ptr)ldsdst, 0, 0);
#else
  *(bf16x8*)ldsdst = *(const bf16x8*)gsrc;
#endif
}
__device__ static inline void async_join() {
#ifdef USE_ASYNC_LDS
  __builtin_amdgcn_s_wait_asynccnt(0);
#endif
}

// A fragment: 16x32 bf16, rows in lanes (lane%16), K pattern per ISA 7.12.2
__device__ static inline bf16x16 load_afrag(const bf16_t* tile, int lr, int lh, int ks) {
  bf16x8 a0 = *(const bf16x8*)(tile + lr * LDT + ks * 32 + lh * 8);
  bf16x8 a1 = *(const bf16x8*)(tile + lr * LDT + ks * 32 + 16 + lh * 8);
  bf16x16 r;
#pragma unroll
  for (int i = 0; i < 8; ++i) { r[i] = a0[i]; r[i + 8] = a1[i]; }
  return r;
}
// B fragment: 32x16 bf16 from LDS tile stored transposed [n][k]
__device__ static inline bf16x16 load_bfrag(const bf16_t* tile, int lr, int lh, int ks) {
  bf16x8 b0 = *(const bf16x8*)(tile + lr * LDT + ks * 32 + lh * 8);
  bf16x8 b1 = *(const bf16x8*)(tile + lr * LDT + ks * 32 + 16 + lh * 8);
  bf16x16 r;
#pragma unroll
  for (int i = 0; i < 8; ++i) { r[i] = b0[i]; r[i + 8] = b1[i]; }
  return r;
}

// ---------------- small utility kernels ----------------
__global__ void k_embed(const int* __restrict__ idx, const float* __restrict__ wte,
                        const float* __restrict__ wpe, float* __restrict__ x) {
  int t = blockIdx.x;
  int tok = idx[t];
  int pos = t & (TSEQ - 1);
  for (int d = threadIdx.x; d < DMODEL; d += 256)
    x[(long)t * DMODEL + d] = wte[(long)tok * DMODEL + d] + wpe[(long)pos * DMODEL + d];
}

__global__ void k_cvt(const float* __restrict__ src, bf16_t* __restrict__ dst, long n) {
  long i = (long)blockIdx.x * 256 + threadIdx.x;
  if (i < n) dst[i] = (bf16_t)src[i];
}

__global__ void k_zero(float* __restrict__ p, long n) {
  long i = (long)blockIdx.x * 256 + threadIdx.x;
  if (i < n) p[i] = 0.f;
}

// layernorm: f32 in, bf16 out
__global__ __launch_bounds__(256) void k_ln(const float* __restrict__ x,
                                            const float* __restrict__ w,
                                            const float* __restrict__ b,
                                            bf16_t* __restrict__ out) {
  __shared__ float red[256];
  int t = blockIdx.x;
  const float* xr = x + (long)t * DMODEL;
  float s = 0.f;
  for (int d = threadIdx.x; d < DMODEL; d += 256) s += xr[d];
  red[threadIdx.x] = s; __syncthreads();
  for (int o = 128; o > 0; o >>= 1) { if (threadIdx.x < o) red[threadIdx.x] += red[threadIdx.x + o]; __syncthreads(); }
  float mu = red[0] / (float)DMODEL;
  __syncthreads();
  float v = 0.f;
  for (int d = threadIdx.x; d < DMODEL; d += 256) { float c = xr[d] - mu; v += c * c; }
  red[threadIdx.x] = v; __syncthreads();
  for (int o = 128; o > 0; o >>= 1) { if (threadIdx.x < o) red[threadIdx.x] += red[threadIdx.x + o]; __syncthreads(); }
  float rstd = rsqrtf(red[0] / (float)DMODEL + 1e-5f);
  for (int d = threadIdx.x; d < DMODEL; d += 256)
    out[(long)t * DMODEL + d] = (bf16_t)((xr[d] - mu) * rstd * w[d] + b[d]);
}

// ---------------- generic WMMA GEMM ----------------
// C[M,N] = act( A[M,K](bf16) * B + bias ) (+ res); per-z batch strides (elements).
// TRANSB=1: B is [N,K] row-major.  TRANSB=0: B is [K,N].
// B_BF16: B already bf16 (raw async staging).  C_BF16: store C as bf16.
// act: 0=none 1=gelu(tanh) 2=gelu(exact)
#define BM 64
#define BN 128
#define BK 64
template <int TRANSB, int B_BF16, int C_BF16>
__global__ __launch_bounds__(256)
void k_gemm(const bf16_t* __restrict__ A, const void* __restrict__ Bp,
            const float* __restrict__ bias, const float* __restrict__ res,
            void* __restrict__ Cp, int M, int N, int Kd, int act,
            long sA, long sB, long sC, long sBias) {
  __shared__ __align__(16) bf16_t As[BM * LDT];
  __shared__ __align__(16) bf16_t Bs[BN * LDT];

  const int z = blockIdx.z;
  const bf16_t* Ab = A + (long)z * sA;
  const float* biasb = bias ? bias + (long)z * sBias : nullptr;
  const float* resb  = res  ? res  + (long)z * sC    : nullptr;

  const int bm = blockIdx.y * BM;
  const int bn = blockIdx.x * BN;
  const int tid = threadIdx.x;
  const int lane = tid & 31, wave = tid >> 5;
  const int lr = lane & 15, lh = lane >> 4;

  f32x8 acc[4];
#pragma unroll
  for (int i = 0; i < 4; ++i)
#pragma unroll
    for (int v = 0; v < 8; ++v) acc[i][v] = 0.f;

  for (int k0 = 0; k0 < Kd; k0 += BK) {
    // ---- stage A (BM x BK bf16): raw 16B chunks, async when available ----
#pragma unroll
    for (int rep = 0; rep < 2; ++rep) {
      int cid = rep * 256 + tid;           // 512 chunks
      int r = cid >> 3, c8 = (cid & 7) * 8;
      copy16_g2l(&As[r * LDT + c8], Ab + (long)(bm + r) * Kd + k0 + c8);
    }
    // ---- stage B (BN x BK) into [n][k] ----
    if (B_BF16) {          // bf16 [N,K] weights: raw async chunks
      const bf16_t* Bb = (const bf16_t*)Bp + (long)z * sB;
#pragma unroll
      for (int rep = 0; rep < 4; ++rep) {
        int cid = rep * 256 + tid;         // 1024 chunks
        int n = cid >> 3, c8 = (cid & 7) * 8;
        if (bn + n < N) {
          copy16_g2l(&Bs[n * LDT + c8], Bb + (long)(bn + n) * Kd + k0 + c8);
        } else {
          bf16x8 zz;
#pragma unroll
          for (int u = 0; u < 8; ++u) zz[u] = (bf16_t)0.f;
          *(bf16x8*)&Bs[n * LDT + c8] = zz;
        }
      }
    } else if (TRANSB) {   // f32 [N,K] weights: load + convert
      const float* Bb = (const float*)Bp + (long)z * sB;
#pragma unroll
      for (int rep = 0; rep < 8; ++rep) {
        int lin = (rep * 256 + tid) * 4;
        int n = lin >> 6, c = lin & 63;
        float4 v4;
        if (bn + n < N) {
          v4 = *(const float4*)(Bb + (long)(bn + n) * Kd + k0 + c);
          if (k0 + BK < Kd) __builtin_prefetch(Bb + (long)(bn + n) * Kd + k0 + BK + c, 0, 1);
        } else { v4.x = v4.y = v4.z = v4.w = 0.f; }
        bf16_t* dst = &Bs[n * LDT + c];
        dst[0] = (bf16_t)v4.x; dst[1] = (bf16_t)v4.y; dst[2] = (bf16_t)v4.z; dst[3] = (bf16_t)v4.w;
      }
    } else {               // f32 [K,N] weights: load + convert + transpose
      const float* Bb = (const float*)Bp + (long)z * sB;
#pragma unroll
      for (int rep = 0; rep < 8; ++rep) {
        int lin = (rep * 256 + tid) * 4;
        int kk = lin >> 7, n = lin & 127;
        float vv[4];
        if (bn + n + 3 < N) {
          float4 v4 = *(const float4*)(Bb + (long)(k0 + kk) * N + bn + n);
          vv[0] = v4.x; vv[1] = v4.y; vv[2] = v4.z; vv[3] = v4.w;
        } else {
#pragma unroll
          for (int u = 0; u < 4; ++u) {
            int nn = bn + n + u;
            vv[u] = (nn < N) ? Bb[(long)(k0 + kk) * N + nn] : 0.f;
          }
        }
#pragma unroll
        for (int u = 0; u < 4; ++u) Bs[(n + u) * LDT + kk] = (bf16_t)vv[u];
      }
    }
    async_join();
    __syncthreads();

    // ---- compute: wave owns 16 N-columns, 4 M-subtiles ----
#pragma unroll
    for (int ks = 0; ks < 2; ++ks) {
      bf16x16 bfm = load_bfrag(&Bs[(wave * 16) * LDT], lr, lh, ks);
#pragma unroll
      for (int mi = 0; mi < 4; ++mi) {
        bf16x16 af = load_afrag(&As[(mi * 16) * LDT], lr, lh, ks);
        acc[mi] = __builtin_amdgcn_wmma_f32_16x16x32_bf16(
            false, af, false, bfm, (short)0, acc[mi], false, false);
      }
    }
    __syncthreads();
  }

  // ---- epilogue ----
  const int n = bn + wave * 16 + lr;
  if (n < N) {
    float bv = biasb ? biasb[n] : 0.f;
#pragma unroll
    for (int mi = 0; mi < 4; ++mi) {
#pragma unroll
      for (int v = 0; v < 8; ++v) {
        int m = bm + mi * 16 + v + 8 * lh;
        if (m < M) {
          float x = acc[mi][v] + bv;
          if (act == 1) {
            float x3 = x * x * x;
            x = 0.5f * x * (1.f + tanhf(0.7978845608028654f * (x + 0.044715f * x3)));
          } else if (act == 2) {
            x = 0.5f * x * (1.f + erff(x * 0.7071067811865476f));
          }
          if (resb) x += resb[(long)m * N + n];
          if (C_BF16) ((bf16_t*)Cp)[(long)z * sC + (long)m * N + n] = (bf16_t)x;
          else        ((float*) Cp)[(long)z * sC + (long)m * N + n] = x;
        }
      }
    }
  }
}

// ---------------- flash attention (WMMA QK^T and PV), bf16 qkv in / bf16 out ----
// grid: (T/128, H, B); block 256 (8 waves, 16 q-rows per wave)
__global__ __launch_bounds__(256)
void k_attn(const bf16_t* __restrict__ qkv, bf16_t* __restrict__ yout) {
  __shared__ __align__(16) bf16_t Qs[128 * LDT];
  __shared__ __align__(16) bf16_t Ks[64 * LDT];    // [key][d]
  __shared__ __align__(16) bf16_t Vt[64 * LDT];    // [d][key]
  __shared__ __align__(16) bf16_t SP[8 * 16 * LDT];// scores -> probs (in place)
  __shared__ float rowM[128], rowL[128], rowS[128];

  const int tid = threadIdx.x, lane = tid & 31, wave = tid >> 5;
  const int lr = lane & 15, lh = lane >> 4;
  const int qb = blockIdx.x * 128;
  const int h  = blockIdx.y;
  const int b  = blockIdx.z;
  const long base = (long)b * TSEQ * QKV3;

  // stage Q raw (scale applied to score accumulators instead)
#pragma unroll
  for (int rep = 0; rep < 4; ++rep) {
    int cid = rep * 256 + tid;             // 1024 chunks of 16B
    int r = cid >> 3, c8 = (cid & 7) * 8;
    copy16_g2l(&Qs[r * LDT + c8], qkv + base + (long)(qb + r) * QKV3 + h * HDIM + c8);
  }
  if (tid < 128) { rowM[tid] = -1e30f; rowL[tid] = 0.f; }
  async_join();

  f32x8 accy[4];
#pragma unroll
  for (int i = 0; i < 4; ++i)
#pragma unroll
    for (int v = 0; v < 8; ++v) accy[i][v] = 0.f;

  const int kend = qb + 128;
  for (int k0 = 0; k0 < kend; k0 += 64) {
    __syncthreads();
    // K rows: raw async chunks; V: transposed element copy
#pragma unroll
    for (int rep = 0; rep < 2; ++rep) {
      int cid = rep * 256 + tid;           // 512 chunks
      int r = cid >> 3, c8 = (cid & 7) * 8;
      copy16_g2l(&Ks[r * LDT + c8], qkv + base + (long)(k0 + r) * QKV3 + DMODEL + h * HDIM + c8);
    }
    for (int i = tid; i < 64 * HDIM; i += 256) {
      int r = i >> 6, d = i & 63;
      Vt[d * LDT + r] = qkv[base + (long)(k0 + r) * QKV3 + 2 * DMODEL + h * HDIM + d];
    }
    async_join();
    __syncthreads();

    // scores = (Q*K^T) * 0.125, masked -> SP
#pragma unroll
    for (int ni = 0; ni < 4; ++ni) {
      f32x8 s;
#pragma unroll
      for (int v = 0; v < 8; ++v) s[v] = 0.f;
#pragma unroll
      for (int ks = 0; ks < 2; ++ks) {
        bf16x16 af = load_afrag(&Qs[(wave * 16) * LDT], lr, lh, ks);
        bf16x16 bfm = load_bfrag(&Ks[(ni * 16) * LDT], lr, lh, ks);
        s = __builtin_amdgcn_wmma_f32_16x16x32_bf16(false, af, false, bfm, (short)0, s, false, false);
      }
      int col = ni * 16 + lr, kg = k0 + col;
#pragma unroll
      for (int v = 0; v < 8; ++v) {
        int rloc = v + 8 * lh;
        int qg = qb + wave * 16 + rloc;
        SP[(wave * 16 + rloc) * LDT + col] = (bf16_t)((kg <= qg) ? s[v] * 0.125f : -1e30f);
      }
    }
    __syncthreads();

    // online-softmax stats: one lane per q-row
    if (lane < 16) {
      int row = wave * 16 + lane;
      float mo = rowM[row], mx = mo;
      for (int c = 0; c < 64; ++c) mx = fmaxf(mx, (float)SP[row * LDT + c]);
      float sc = __expf(mo - mx);
      float l = rowL[row] * sc;
      for (int c = 0; c < 64; ++c) {
        float sv = (float)SP[row * LDT + c];
        float e = (sv <= -1e29f) ? 0.f : __expf(sv - mx);
        l += e;
        SP[row * LDT + c] = (bf16_t)e;
      }
      rowM[row] = mx; rowL[row] = l; rowS[row] = sc;
    }
    __syncthreads();

    // rescale accumulators, then accy += P * V
#pragma unroll
    for (int ni = 0; ni < 4; ++ni)
#pragma unroll
      for (int v = 0; v < 8; ++v) accy[ni][v] *= rowS[wave * 16 + v + 8 * lh];
#pragma unroll
    for (int ni = 0; ni < 4; ++ni) {
#pragma unroll
      for (int ks = 0; ks < 2; ++ks) {
        bf16x16 af = load_afrag(&SP[(wave * 16) * LDT], lr, lh, ks);
        bf16x16 bfm = load_bfrag(&Vt[(ni * 16) * LDT], lr, lh, ks);
        accy[ni] = __builtin_amdgcn_wmma_f32_16x16x32_bf16(false, af, false, bfm, (short)0, accy[ni], false, false);
      }
    }
  }
  __syncthreads();

#pragma unroll
  for (int ni = 0; ni < 4; ++ni)
#pragma unroll
    for (int v = 0; v < 8; ++v) {
      int rloc = wave * 16 + v + 8 * lh;
      float o = accy[ni][v] / rowL[rloc];
      int t = qb + rloc;
      yout[((long)b * TSEQ + t) * DMODEL + h * HDIM + ni * 16 + lr] = (bf16_t)o;
    }
}

// ---------------- MoE ----------------
__global__ void k_rlogits(const bf16_t* __restrict__ y, const float* __restrict__ wg,
                          float* __restrict__ rlog) {
  int g = blockIdx.x * 256 + threadIdx.x;
  if (g >= NTOK * NEXP) return;
  int t = g >> 3, e = g & 7;
  const bf16_t* yr = y + (long)t * DMODEL;
  const float* wr = wg + (long)e * DMODEL;
  float s = 0.f;
  for (int d = 0; d < DMODEL; ++d) s += (float)yr[d] * wr[d];
  rlog[g] = s;
}

__global__ void k_route(const float* __restrict__ rlog, int* __restrict__ topi,
                        float* __restrict__ probs, int* __restrict__ slot) {
  for (int t = threadIdx.x; t < NTOK; t += 256) {
    const float* l = rlog + t * NEXP;
    int i1 = 0; float v1 = l[0];
    for (int e = 1; e < NEXP; ++e) if (l[e] > v1) { v1 = l[e]; i1 = e; }
    int i2 = (i1 == 0) ? 1 : 0; float v2 = l[i2];
    for (int e = 0; e < NEXP; ++e) if (e != i1 && l[e] > v2) { v2 = l[e]; i2 = e; }
    float e2 = __expf(v2 - v1);
    probs[t * 2]     = 1.f / (1.f + e2);
    probs[t * 2 + 1] = e2 / (1.f + e2);
    topi[t * 2] = i1; topi[t * 2 + 1] = i2;
  }
  __syncthreads();
  if (threadIdx.x == 0) {   // k-major, token-order ranking (matches jnp.cumsum)
    int cnt[NEXP];
    for (int e = 0; e < NEXP; ++e) cnt[e] = 0;
    for (int k = 0; k < TOPK; ++k)
      for (int t = 0; t < NTOK; ++t) {
        int e = topi[t * 2 + k];
        int r = cnt[e]++;
        slot[t * 2 + k] = (r < CAPE) ? (e * CAPE + r) : -1;
      }
  }
}

__global__ void k_dispatch(const bf16_t* __restrict__ y, const int* __restrict__ slot,
                           bf16_t* __restrict__ ein) {
  int a = blockIdx.x;
  int s = slot[a];
  if (s < 0) return;
  int t = a >> 1;
  for (int d = threadIdx.x; d < DMODEL; d += 256)
    ein[(long)s * DMODEL + d] = y[(long)t * DMODEL + d];
}

__global__ void k_combine(const bf16_t* __restrict__ eout, const int* __restrict__ slot,
                          const float* __restrict__ probs, float* __restrict__ x) {
  int t = blockIdx.x;
  for (int d = threadIdx.x; d < DMODEL; d += 256) {
    float acc = 0.f;
#pragma unroll
    for (int k = 0; k < TOPK; ++k) {
      int s = slot[t * 2 + k];
      if (s >= 0) acc += probs[t * 2 + k] * (float)eout[(long)s * DMODEL + d];
    }
    x[(long)t * DMODEL + d] += acc;
  }
}

// ---------------- host ----------------
extern "C" void kernel_launch(void* const* d_in, const int* in_sizes, int n_in,
                              void* d_out, int out_size, void* d_ws, size_t ws_size,
                              hipStream_t stream) {
  (void)in_sizes; (void)n_in; (void)out_size; (void)ws_size;
  const int*   idx   = (const int*)d_in[0];
  const float* wte   = (const float*)d_in[1];
  const float* wpe   = (const float*)d_in[2];
  const float* lnfw  = (const float*)d_in[3];
  const float* lnfb  = (const float*)d_in[4];
  const float* ln1w  = (const float*)d_in[5];
  const float* ln1b  = (const float*)d_in[6];
  const float* ln2w  = (const float*)d_in[7];
  const float* ln2b  = (const float*)d_in[8];
  const float* qkvw  = (const float*)d_in[9];
  const float* qkvb  = (const float*)d_in[10];
  const float* projw = (const float*)d_in[11];
  const float* projb = (const float*)d_in[12];
  const float* mfcw  = (const float*)d_in[13];
  const float* mfcb  = (const float*)d_in[14];
  const float* mpjw  = (const float*)d_in[15];
  const float* mpjb  = (const float*)d_in[16];
  const float* rtw   = (const float*)d_in[17];
  const float* efcw  = (const float*)d_in[18];
  const float* efcb  = (const float*)d_in[19];
  const float* epjw  = (const float*)d_in[20];
  const float* epjb  = (const float*)d_in[21];

  char* wsb = (char*)d_ws;
  size_t off = 0;
  auto AL = [](size_t v) { return (v + 255) & ~(size_t)255; };

  float*  X   = (float*)(wsb + off);  off = AL(off + (size_t)NTOK * DMODEL * 4);
  bf16_t* Y   = (bf16_t*)(wsb + off); off = AL(off + (size_t)NTOK * DMODEL * 2);
  bf16_t* ATT = (bf16_t*)(wsb + off); off = AL(off + (size_t)NTOK * DMODEL * 2);
  bf16_t* BIG = (bf16_t*)(wsb + off); off = AL(off + (size_t)NEXP * CAPE * FFD * 2); // QKV / EH
  bf16_t* QKV = BIG;
  bf16_t* EH  = BIG;
  bf16_t* B2  = (bf16_t*)(wsb + off); off = AL(off + (size_t)2 * NEXP * CAPE * DMODEL * 2); // FFH / EIN+EOUT
  bf16_t* FFH = B2;
  bf16_t* EIN = B2;
  bf16_t* EOUT = B2 + (size_t)NEXP * CAPE * DMODEL;
  float*  RLOG = (float*)(wsb + off); off = AL(off + (size_t)NTOK * NEXP * 4);
  float*  PRB  = (float*)(wsb + off); off = AL(off + (size_t)NTOK * TOPK * 4);
  int*    TPI  = (int*)(wsb + off);   off = AL(off + (size_t)NTOK * TOPK * 4);
  int*    SLT  = (int*)(wsb + off);   off = AL(off + (size_t)NTOK * TOPK * 4);
  bf16_t* WTEB = (bf16_t*)(wsb + off); off = AL(off + (size_t)NVOC * DMODEL * 2);

  // one-time (per call): bf16 copy of wte for the tied-head GEMM (fits in 192MB L2)
  {
    long n = (long)NVOC * DMODEL;
    k_cvt<<<(unsigned)((n + 255) / 256), 256, 0, stream>>>(wte, WTEB, n);
  }
  k_embed<<<NTOK, 256, 0, stream>>>(idx, wte, wpe, X);

  for (int i = 0; i < NLAYER; ++i) {
    int jj = (i / 2 < 1) ? (i / 2) : 1;   // JAX clamps OOB index into shape-(2,..) arrays

    // ---- attention ----
    k_ln<<<NTOK, 256, 0, stream>>>(X, ln1w + (long)i * DMODEL, ln1b + (long)i * DMODEL, Y);
    k_gemm<1, 0, 1><<<dim3(QKV3 / BN, NTOK / BM, 1), 256, 0, stream>>>(
        Y, qkvw + (long)i * QKV3 * DMODEL, qkvb + (long)i * QKV3, nullptr,
        QKV, NTOK, QKV3, DMODEL, 0, 0, 0, 0, 0);
    k_attn<<<dim3(TSEQ / 128, NHEAD, 2), 256, 0, stream>>>(QKV, ATT);
    k_gemm<1, 0, 0><<<dim3(DMODEL / BN, NTOK / BM, 1), 256, 0, stream>>>(
        ATT, projw + (long)i * DMODEL * DMODEL, projb + (long)i * DMODEL, X,
        X, NTOK, DMODEL, DMODEL, 0, 0, 0, 0, 0);

    // ---- FFN / MoE ----
    k_ln<<<NTOK, 256, 0, stream>>>(X, ln2w + (long)i * DMODEL, ln2b + (long)i * DMODEL, Y);
    if ((i & 1) == 0) {   // MoE
      k_rlogits<<<(NTOK * NEXP) / 256, 256, 0, stream>>>(Y, rtw + (long)jj * NEXP * DMODEL, RLOG);
      k_route<<<1, 256, 0, stream>>>(RLOG, TPI, PRB, SLT);
      long nz = (long)NEXP * CAPE * DMODEL * 2 / 4;   // bf16 buffer zeroed as f32 words
      k_zero<<<(unsigned)((nz + 255) / 256), 256, 0, stream>>>((float*)EIN, nz);
      k_dispatch<<<NTOK * TOPK, 256, 0, stream>>>(Y, SLT, EIN);
      k_gemm<0, 0, 1><<<dim3(FFD / BN, CAPE / BM, NEXP), 256, 0, stream>>>(
          EIN, efcw + (long)jj * NEXP * DMODEL * FFD, efcb + (long)jj * NEXP * FFD, nullptr,
          EH, CAPE, FFD, DMODEL, 2,
          (long)CAPE * DMODEL, (long)DMODEL * FFD, (long)CAPE * FFD, FFD);
      k_gemm<0, 0, 1><<<dim3(DMODEL / BN, CAPE / BM, NEXP), 256, 0, stream>>>(
          EH, epjw + (long)jj * NEXP * FFD * DMODEL, epjb + (long)jj * NEXP * DMODEL, nullptr,
          EOUT, CAPE, DMODEL, FFD, 0,
          (long)CAPE * FFD, (long)FFD * DMODEL, (long)CAPE * DMODEL, DMODEL);
      k_combine<<<NTOK, 256, 0, stream>>>(EOUT, SLT, PRB, X);
    } else {              // dense MLP (tanh-gelu)
      k_gemm<1, 0, 1><<<dim3(FFD / BN, NTOK / BM, 1), 256, 0, stream>>>(
          Y, mfcw + (long)jj * FFD * DMODEL, mfcb + (long)jj * FFD, nullptr,
          FFH, NTOK, FFD, DMODEL, 1, 0, 0, 0, 0);
      k_gemm<1, 0, 0><<<dim3(DMODEL / BN, NTOK / BM, 1), 256, 0, stream>>>(
          FFH, mpjw + (long)jj * DMODEL * FFD, mpjb + (long)jj * DMODEL, X,
          X, NTOK, DMODEL, FFD, 0, 0, 0, 0, 0);
    }
  }

  // ---- final LN + tied-head logits (bf16 x bf16 -> f32) ----
  k_ln<<<NTOK, 256, 0, stream>>>(X, lnfw, lnfb, Y);
  k_gemm<1, 1, 0><<<dim3((NVOC + BN - 1) / BN, NTOK / BM, 1), 256, 0, stream>>>(
      Y, WTEB, nullptr, nullptr, (float*)d_out, NTOK, NVOC, DMODEL, 0, 0, 0, 0, 0);
}
